// SFNOLayer_7026566496835
// MI455X (gfx1250) — matrix operand order repaired and checked by reading, not compiled, not run on hardware
//
#include <hip/hip_runtime.h>
#include <math.h>

#ifndef M_PI
#define M_PI 3.14159265358979323846
#endif

typedef __attribute__((ext_vector_type(16))) _Float16 v16h;
typedef __attribute__((ext_vector_type(8)))  float    v8f;

#define B_    2
#define CH_   128
#define NLAT_ 256
#define NLON_ 512
#define DM_   512
#define LMAX_ 128

// workspace layout (float offsets)
#define F_P0W    0          // [LMAX][NLAT]  Pbar_l^0(x_k)*w_k  (l-major)
#define F_PRAW   32768      // [LMAX][NLAT]  Pbar_l^0(x_k)
#define F_Q      65536      // [LMAX]        mean_k Pbar_l^0(x_k)
#define F_POOLED 65664      // [B*CH]
#define F_H      65920      // (unused spare)
#define F_DELTA  66944      // [B*CH]
#define F_S2PI   67200      // [B*CH][NLAT]  2pi * lon-mean of x
#define F_C0     132736     // [B*CH][LMAX]  m=0 SHT coeffs

// ---------------------------------------------------------------------------
// Constants: Clenshaw-Curtis weights + fully-normalized Legendre P_l^0, in f64
// (runs once per call, 256 threads, negligible). Also zeroes the pooled accum.
// ---------------------------------------------------------------------------
__global__ void k_constants(float* ws) {
  int k = threadIdx.x;                      // 0..255 (latitude node)
  double theta = M_PI * (double)k / 255.0;  // N = NLAT-1 = 255
  double xk = cos(theta);
  double s = 0.0;
  for (int j = 1; j <= 127; ++j)            // N odd -> b_j == 2 always
    s += (2.0 / (4.0 * j * j - 1.0)) * cos(2.0 * j * theta);
  double w = (2.0 / 255.0) * (1.0 - s);
  if (k == 0 || k == 255) w *= 0.5;

  double pmm = sqrt(1.0 / (4.0 * M_PI));    // m = 0
  double p0 = pmm;
  double p1 = sqrt(3.0) * xk * pmm;
  ws[F_P0W  + 0 * NLAT_ + k] = (float)(p0 * w);
  ws[F_PRAW + 0 * NLAT_ + k] = (float)p0;
  ws[F_P0W  + 1 * NLAT_ + k] = (float)(p1 * w);
  ws[F_PRAW + 1 * NLAT_ + k] = (float)p1;
  for (int l = 2; l < LMAX_; ++l) {
    double ld = (double)l;
    double a  = sqrt((4.0 * ld * ld - 1.0) / (ld * ld));
    double bb = sqrt((2.0 * ld + 1.0) * (ld - 1.0) * (ld - 1.0) /
                     ((ld * ld) * (2.0 * ld - 3.0)));
    double p2 = a * xk * p1 - bb * p0;
    ws[F_P0W  + l * NLAT_ + k] = (float)(p2 * w);
    ws[F_PRAW + l * NLAT_ + k] = (float)p2;
    p0 = p1; p1 = p2;
  }
  ws[F_POOLED + k] = 0.0f;                  // zero atomic accumulator (B*CH=256)
}

__global__ void k_qmean(float* ws) {
  int l = threadIdx.x;                      // 0..127
  float s = 0.f;
  for (int k = 0; k < NLAT_; ++k) s += ws[F_PRAW + l * NLAT_ + k];
  ws[F_Q + l] = s * (1.0f / (float)NLAT_);
}

// ---------------------------------------------------------------------------
// Longitude mean (rfft bin 0, norm='forward', x 2pi). One wave per row,
// fully-coalesced float4 loads, shfl_xor wave32 reduction.
// ---------------------------------------------------------------------------
__global__ void k_lonmean(const float* __restrict__ x, float* __restrict__ ws) {
  int lane = threadIdx.x & 31;
  int wv   = threadIdx.x >> 5;
  int row  = blockIdx.x * 8 + wv;           // 0 .. B*CH*NLAT-1 = 65535
  const float4* xr = (const float4*)(x + (size_t)row * NLON_);
  float s = 0.f;
#pragma unroll
  for (int t = 0; t < 4; ++t) {
    float4 v = xr[lane + 32 * t];
    s += v.x + v.y + v.z + v.w;
  }
#pragma unroll
  for (int off = 16; off > 0; off >>= 1) s += __shfl_xor(s, off, 32);
  if (lane == 0)
    ws[F_S2PI + row] = s * (float)(2.0 * M_PI / (double)NLON_);
}

// ---------------------------------------------------------------------------
// m=0 SHT:  C0[M=256][L=128] = S2PI[256][256] * P0W[l][k]^T.
// One wave per 16x16 tile; V_WMMA_F32_16X16X32_F16 with hi/lo f16 split
// (AhBh + AlBh + AhBl) for ~f32 fidelity into the f32 accumulator.
// ---------------------------------------------------------------------------
__global__ __launch_bounds__(32) void k_sht_gemm(float* ws) {
  const float* A  = ws + F_S2PI;            // [256][256]
  const float* Bm = ws + F_P0W;             // [128][256] (l-major)
  float*       C  = ws + F_C0;              // [256][128]
  int lane  = threadIdx.x;
  int tileM = blockIdx.x & 15;              // 16 M-tiles
  int tileN = blockIdx.x >> 4;              // 8 N-tiles (l)
  int r     = lane & 15;
  int hsel  = (lane >> 4) & 1;

  v8f acc = {};
  for (int kb = 0; kb < 8; ++kb) {
    int k0 = kb * 32;
    v16h ah, al, bh, bl;
    // A 16x32 f16 layout (ISA 7.12.2): lane<16 holds K {0..7,16..23}, lane>=16 K {8..15,24..31}
#pragma unroll
    for (int j = 0; j < 16; ++j) {
      int kk = k0 + ((j < 8) ? (hsel * 8 + j) : (16 + hsel * 8 + (j - 8)));
      float av = A[(tileM * 16 + r) * NLAT_ + kk];
      _Float16 h = (_Float16)av;
      ah[j] = h;
      al[j] = (_Float16)(av - (float)h);
    }
    // B 32x16 f16 layout: lane<16 holds K {0..7,16..23}, lane>=16 K {8..15,24..31}; N = lane&15
#pragma unroll
    for (int e = 0; e < 16; ++e) {
      int kk = k0 + ((e < 8) ? (hsel * 8 + e) : (16 + hsel * 8 + (e - 8)));
      float bv = Bm[(tileN * 16 + r) * NLAT_ + kk];
      _Float16 h = (_Float16)bv;
      bh[e] = h;
      bl[e] = (_Float16)(bv - (float)h);
    }
    acc = __builtin_amdgcn_wmma_f32_16x16x32_f16(false, ah, false, bh, (short)0, acc, false, false);
    acc = __builtin_amdgcn_wmma_f32_16x16x32_f16(false, al, false, bh, (short)0, acc, false, false);
    acc = __builtin_amdgcn_wmma_f32_16x16x32_f16(false, ah, false, bl, (short)0, acc, false, false);
  }
  // C/D f32 layout: VGPR v, lane<16 -> M=v, N=lane; lane>=16 -> M=v+8, N=lane-16
#pragma unroll
  for (int v = 0; v < 8; ++v) {
    int m = tileM * 16 + v + hsel * 8;
    int n = tileN * 16 + r;
    C[m * LMAX_ + n] = acc[v];
  }
}

// ---------------------------------------------------------------------------
// pooled[b,c] = sum_i sum_l C0[b,i,l] * fr[i,c,l] * q[l]
// (filter_imag cancels: m=0 SHT coeffs are purely real, only Re is pooled)
// ---------------------------------------------------------------------------
__global__ void k_pooled(const float* __restrict__ fr, float* ws) {
  int b  = blockIdx.x >> 4;
  int i0 = (blockIdx.x & 15) * 8;
  int c  = threadIdx.x;                     // 0..127
  const float* q = ws + F_Q;
  float acc = 0.f;
  for (int ii = 0; ii < 8; ++ii) {
    int i = i0 + ii;
    const float* c0row = ws + F_C0 + (size_t)(b * CH_ + i) * LMAX_;
    const float* frrow = fr + ((size_t)i * CH_ + c) * LMAX_;
#pragma unroll 4
    for (int l = 0; l < LMAX_; ++l) acc += c0row[l] * frrow[l] * q[l];
  }
  atomicAdd(&ws[F_POOLED + b * CH_ + c], acc);
}

// ---------------------------------------------------------------------------
// MLP: h = gelu_exact(pooled @ w1 + b1); delta = h @ w2 + b2
// ---------------------------------------------------------------------------
__global__ void k_mlp(const float* __restrict__ w1, const float* __restrict__ b1,
                      const float* __restrict__ w2, const float* __restrict__ b2,
                      float* ws) {
  __shared__ float pl[CH_];
  __shared__ float hl[DM_];
  int b = blockIdx.x;
  int d = threadIdx.x;                      // 0..511
  if (d < CH_) pl[d] = ws[F_POOLED + b * CH_ + d];
  __syncthreads();
  float z = b1[d];
  for (int cc = 0; cc < CH_; ++cc) z += pl[cc] * w1[cc * DM_ + d];
  hl[d] = 0.5f * z * (1.0f + erff(z * 0.70710678118654752f));   // exact GELU
  __syncthreads();
  if (d < CH_) {
    float z2 = b2[d];
    for (int dd = 0; dd < DM_; ++dd) z2 += hl[dd] * w2[dd * CH_ + d];
    ws[F_DELTA + b * CH_ + d] = z2;
  }
}

// ---------------------------------------------------------------------------
// out = x + delta[b,c]  (float4, second x read hits 192MB L2)
// ---------------------------------------------------------------------------
__global__ void k_residual(const float* __restrict__ x, const float* __restrict__ ws,
                           float* __restrict__ out) {
  size_t idx = (size_t)blockIdx.x * blockDim.x + threadIdx.x;   // float4 index
  const float4* x4 = (const float4*)x;
  float4* o4 = (float4*)out;
  float d = ws[F_DELTA + (int)(idx >> 15)];   // NLAT*NLON/4 = 32768 float4 per (b,c)
  float4 v = x4[idx];
  v.x += d; v.y += d; v.z += d; v.w += d;
  o4[idx] = v;
}

extern "C" void kernel_launch(void* const* d_in, const int* in_sizes, int n_in,
                              void* d_out, int out_size, void* d_ws, size_t ws_size,
                              hipStream_t stream) {
  (void)in_sizes; (void)n_in; (void)out_size; (void)ws_size;
  const float* x  = (const float*)d_in[0];
  const float* fr = (const float*)d_in[1];
  // d_in[2] = filter_imag: provably unused (only Re of purely-real m=0 coeffs survives pooling)
  const float* w1 = (const float*)d_in[3];
  const float* b1 = (const float*)d_in[4];
  const float* w2 = (const float*)d_in[5];
  const float* b2 = (const float*)d_in[6];
  float* ws  = (float*)d_ws;
  float* out = (float*)d_out;

  hipLaunchKernelGGL(k_constants, dim3(1),     dim3(256), 0, stream, ws);
  hipLaunchKernelGGL(k_qmean,     dim3(1),     dim3(128), 0, stream, ws);
  hipLaunchKernelGGL(k_lonmean,   dim3(8192),  dim3(256), 0, stream, x, ws);
  hipLaunchKernelGGL(k_sht_gemm,  dim3(128),   dim3(32),  0, stream, ws);
  hipLaunchKernelGGL(k_pooled,    dim3(32),    dim3(128), 0, stream, fr, ws);
  hipLaunchKernelGGL(k_mlp,       dim3(2),     dim3(512), 0, stream, w1, b1, w2, b2, ws);
  hipLaunchKernelGGL(k_residual,  dim3(32768), dim3(256), 0, stream, x, ws, out);
}
